// ResidualBlock_61847529062596
// MI455X (gfx1250) — matrix operand order repaired and checked by reading, not compiled, not run on hardware
//
#include <hip/hip_runtime.h>
#include <hip/hip_bf16.h>
#include <math.h>

// ---------------- constants (match reference) ----------------
#define D_MODEL 1024
#define D_INNER 2048
#define D_STATE 16
#define DT_RANK 64
#define D_CONV  4
#define SEQ_L   2048
#define LN_EPS  1e-5f

typedef __attribute__((ext_vector_type(16))) __bf16 v16bf;
typedef __attribute__((ext_vector_type(8)))  __bf16 v8bf;
typedef __attribute__((ext_vector_type(8)))  float  v8f;

// ---------------- CDNA5 async global->LDS helpers ----------------
// GLOBAL_LOAD_ASYNC_TO_LDS_B32 (GV mode): VDST = LDS byte address (VGPR),
// VADDR = 64-bit global address.  Tracked by ASYNCcnt.
__device__ __forceinline__ void async_load_f32_to_lds(unsigned lds_off,
                                                      const float* gptr) {
  asm volatile("global_load_async_to_lds_b32 %0, %1, off"
               :: "v"(lds_off), "v"(gptr)
               : "memory");
}
__device__ __forceinline__ void wait_asynccnt0() {
  asm volatile("s_wait_asynccnt 0x0" ::: "memory");
}

// ---------------- WMMA fragment loaders ----------------
// A: M x K row-major (bf16). A fragment, contiguous k order used here (same as B):
//   lanes 0-15  hold K k0+0..15 of row m0+lane
//   lanes 16-31 hold K k0+16..31 of row m0+lane-16
__device__ __forceinline__ v16bf load_a_frag(const __bf16* __restrict__ Arow,
                                             int k0, int half) {
  const int ka = k0 + half * 16;
  v8bf lo = *(const v8bf*)(Arow + ka);
  v8bf hi = *(const v8bf*)(Arow + ka + 8);
  v16bf a;
#pragma unroll
  for (int i = 0; i < 8; ++i) { a[i] = lo[i]; a[i + 8] = hi[i]; }
  return a;
}
// Bt: N x K row-major (B transposed). B fragment, contiguous k order used here (same as A):
//   lanes 0-15 hold K k0+0..15 of column n0+lane, lanes 16-31 hold K k0+16..31.
__device__ __forceinline__ v16bf load_b_frag(const __bf16* __restrict__ Brow,
                                             int k0, int half) {
  return *(const v16bf*)(Brow + k0 + half * 16);
}

// ---------------- 1x1 tile (used only for the narrow N=96 GEMM) -------------
__device__ __forceinline__ v8f wmma_tile_bf16(const __bf16* __restrict__ A,
                                              const __bf16* __restrict__ Bt,
                                              int K, int m0, int n0) {
  const int lane = threadIdx.x & 31;
  const int half = lane >> 4;
  const int l15  = lane & 15;
  const __bf16* Arow = A  + (size_t)(m0 + l15) * (size_t)K;
  const __bf16* Brow = Bt + (size_t)(n0 + l15) * (size_t)K;
  v8f c = {};
  for (int k0 = 0; k0 < K; k0 += 32) {
    __builtin_prefetch((const void*)(Arow + k0 + 128), 0, 1);
    __builtin_prefetch((const void*)(Brow + k0 + 128), 0, 1);
    v16bf a = load_a_frag(Arow, k0, half);
    v16bf b = load_b_frag(Brow, k0, half);
    c = __builtin_amdgcn_wmma_f32_16x16x32_bf16(false, a, false, b,
                                                (short)0, c, false, false);
  }
  return c;
}

// ---------------- 2x4 register-blocked wave GEMM (32x64 output per wave) ----
// 0.75 fragment-loads per wmma; 8 back-to-back WMMAs per K-step.
__device__ __forceinline__ void wmma_block_2x4(const __bf16* __restrict__ A,
                                               const __bf16* __restrict__ Bt,
                                               int K, int m0, int n0,
                                               v8f acc[2][4]) {
  const int lane = threadIdx.x & 31;
  const int half = lane >> 4;
  const int l15  = lane & 15;
  const __bf16* Ar0 = A + (size_t)(m0 + l15) * (size_t)K;
  const __bf16* Ar1 = Ar0 + (size_t)16 * (size_t)K;
  const __bf16* Br0 = Bt + (size_t)(n0 + l15) * (size_t)K;
  const __bf16* Br1 = Br0 + (size_t)16 * (size_t)K;
  const __bf16* Br2 = Br0 + (size_t)32 * (size_t)K;
  const __bf16* Br3 = Br0 + (size_t)48 * (size_t)K;
#pragma unroll
  for (int i = 0; i < 2; ++i)
#pragma unroll
    for (int j = 0; j < 4; ++j) acc[i][j] = (v8f){};

  for (int k0 = 0; k0 < K; k0 += 32) {
    __builtin_prefetch((const void*)(Ar0 + k0 + 128), 0, 1);
    __builtin_prefetch((const void*)(Ar1 + k0 + 128), 0, 1);
    __builtin_prefetch((const void*)(Br0 + k0 + 128), 0, 1);
    __builtin_prefetch((const void*)(Br2 + k0 + 128), 0, 1);
    v16bf a0 = load_a_frag(Ar0, k0, half);
    v16bf a1 = load_a_frag(Ar1, k0, half);
    v16bf b0 = load_b_frag(Br0, k0, half);
    v16bf b1 = load_b_frag(Br1, k0, half);
    v16bf b2 = load_b_frag(Br2, k0, half);
    v16bf b3 = load_b_frag(Br3, k0, half);
    acc[0][0] = __builtin_amdgcn_wmma_f32_16x16x32_bf16(false, a0, false, b0, (short)0, acc[0][0], false, false);
    acc[1][0] = __builtin_amdgcn_wmma_f32_16x16x32_bf16(false, a1, false, b0, (short)0, acc[1][0], false, false);
    acc[0][1] = __builtin_amdgcn_wmma_f32_16x16x32_bf16(false, a0, false, b1, (short)0, acc[0][1], false, false);
    acc[1][1] = __builtin_amdgcn_wmma_f32_16x16x32_bf16(false, a1, false, b1, (short)0, acc[1][1], false, false);
    acc[0][2] = __builtin_amdgcn_wmma_f32_16x16x32_bf16(false, a0, false, b2, (short)0, acc[0][2], false, false);
    acc[1][2] = __builtin_amdgcn_wmma_f32_16x16x32_bf16(false, a1, false, b2, (short)0, acc[1][2], false, false);
    acc[0][3] = __builtin_amdgcn_wmma_f32_16x16x32_bf16(false, a0, false, b3, (short)0, acc[0][3], false, false);
    acc[1][3] = __builtin_amdgcn_wmma_f32_16x16x32_bf16(false, a1, false, b3, (short)0, acc[1][3], false, false);
  }
}

// wave -> 32x64 block coordinates; false if whole wave out of range
__device__ __forceinline__ bool block_coords_2x4(int M, int N, int& m0, int& n0) {
  const int wave    = blockIdx.x * (blockDim.x >> 5) + (threadIdx.x >> 5);
  const int blocksN = N >> 6;               // 64-wide blocks
  const int bm = wave / blocksN;
  const int bn = wave - bm * blocksN;
  if (bm >= (M >> 5)) return false;
  m0 = bm << 5; n0 = bn << 6;
  return true;
}

// ---------------- kernels ----------------

// LayerNorm (reference formula: var = E[x^2]-mu^2), fp32 in -> bf16 out
__global__ void ln_kernel(const float* __restrict__ x,
                          const float* __restrict__ gamma,
                          const float* __restrict__ beta,
                          __bf16* __restrict__ xn) {
  __shared__ float s1[256];
  __shared__ float s2[256];
  const int row = blockIdx.x;
  const int tid = threadIdx.x;
  const float* xr = x + (size_t)row * D_MODEL;
  float a0 = xr[tid], a1 = xr[tid + 256], a2 = xr[tid + 512], a3 = xr[tid + 768];
  s1[tid] = a0 + a1 + a2 + a3;
  s2[tid] = a0 * a0 + a1 * a1 + a2 * a2 + a3 * a3;
  __syncthreads();
  for (int o = 128; o > 0; o >>= 1) {
    if (tid < o) { s1[tid] += s1[tid + o]; s2[tid] += s2[tid + o]; }
    __syncthreads();
  }
  const float mu  = s1[0] * (1.0f / D_MODEL);
  const float var = s2[0] * (1.0f / D_MODEL) - mu * mu;
  const float rs  = rsqrtf(var + LN_EPS);
  __bf16* orow = xn + (size_t)row * D_MODEL;
#pragma unroll
  for (int j = 0; j < 4; ++j) {
    const int i = tid + j * 256;
    const float v = (xr[i] - mu) * rs * gamma[i] + beta[i];
    orow[i] = (__bf16)v;
  }
}

// transpose + fp32->bf16:  W (K x N) -> Wt (N x K)
__global__ void transpose_bf16_kernel(const float* __restrict__ W,
                                      __bf16* __restrict__ Wt, int K, int N) {
  const int idx = blockIdx.x * blockDim.x + threadIdx.x;
  if (idx >= K * N) return;
  const int k = idx / N;
  const int n = idx - k * N;
  Wt[(size_t)n * K + k] = (__bf16)W[idx];
}

// Aneg = -exp(A_log)
__global__ void aneg_kernel(const float* __restrict__ Alog,
                            float* __restrict__ Aneg, int n) {
  const int idx = blockIdx.x * blockDim.x + threadIdx.x;
  if (idx < n) Aneg[idx] = -__expf(Alog[idx]);
}

// GEMM1: xn_bf (L x D_MODEL) @ W_in -> xr fp32 (L x 2*D_INNER)
__global__ void gemm_in_kernel(const __bf16* __restrict__ A,
                               const __bf16* __restrict__ Bt,
                               float* __restrict__ C) {
  int m0, n0;
  if (!block_coords_2x4(SEQ_L, 2 * D_INNER, m0, n0)) return;
  v8f acc[2][4];
  wmma_block_2x4(A, Bt, D_MODEL, m0, n0, acc);
  const int lane = threadIdx.x & 31;
  const int roff = (lane >> 4) * 8;
  const int l15  = lane & 15;
#pragma unroll
  for (int i = 0; i < 2; ++i)
#pragma unroll
    for (int j = 0; j < 4; ++j) {
      const int col = n0 + 16 * j + l15;
#pragma unroll
      for (int r = 0; r < 8; ++r)
        C[(size_t)(m0 + 16 * i + roff + r) * (2 * D_INNER) + col] = acc[i][j][r];
    }
}

// depthwise causal conv (width 4) + SiLU: xr[:, :D_INNER] -> u2 bf16
__global__ void conv_silu_kernel(const float* __restrict__ xr,
                                 const float* __restrict__ Wc,
                                 const float* __restrict__ bc,
                                 __bf16* __restrict__ u2) {
  const int idx = blockIdx.x * blockDim.x + threadIdx.x;
  if (idx >= SEQ_L * D_INNER) return;
  const int t = idx / D_INNER;
  const int d = idx - t * D_INNER;
  float acc = bc[d];
#pragma unroll
  for (int k = 0; k < D_CONV; ++k) {
    const int tt = t - (D_CONV - 1) + k;
    if (tt >= 0) acc = fmaf(xr[(size_t)tt * (2 * D_INNER) + d], Wc[k * D_INNER + d], acc);
  }
  const float s = acc / (1.0f + __expf(-acc));
  u2[idx] = (__bf16)s;
}

// GEMM2: u2_bf (L x D_INNER) @ W_x -> x_dbl fp32 (L x 96); cols<64 also to dt_bf
__global__ void gemm_xdbl_kernel(const __bf16* __restrict__ A,
                                 const __bf16* __restrict__ Bt,
                                 float* __restrict__ xdbl,
                                 __bf16* __restrict__ dtbf) {
  const int NC = DT_RANK + 2 * D_STATE; // 96
  const int wave   = blockIdx.x * (blockDim.x >> 5) + (threadIdx.x >> 5);
  const int tilesN = NC >> 4;
  const int tm = wave / tilesN;
  const int tn = wave - tm * tilesN;
  if (tm >= (SEQ_L >> 4)) return;
  const int m0 = tm << 4, n0 = tn << 4;
  v8f c = wmma_tile_bf16(A, Bt, D_INNER, m0, n0);
  const int lane = threadIdx.x & 31;
  const int roff = (lane >> 4) * 8;
  const int col  = n0 + (lane & 15);
#pragma unroll
  for (int r = 0; r < 8; ++r) {
    const int row = m0 + roff + r;
    xdbl[(size_t)row * NC + col] = c[r];
    if (col < DT_RANK) dtbf[(size_t)row * DT_RANK + col] = (__bf16)c[r];
  }
}

// GEMM3: dt_bf (L x DT_RANK) @ W_dt + b_dt, softplus -> delta fp32 (L x D_INNER)
__global__ void gemm_delta_kernel(const __bf16* __restrict__ A,
                                  const __bf16* __restrict__ Bt,
                                  const float* __restrict__ bdt,
                                  float* __restrict__ delta) {
  int m0, n0;
  if (!block_coords_2x4(SEQ_L, D_INNER, m0, n0)) return;
  v8f acc[2][4];
  wmma_block_2x4(A, Bt, DT_RANK, m0, n0, acc);
  const int lane = threadIdx.x & 31;
  const int roff = (lane >> 4) * 8;
  const int l15  = lane & 15;
#pragma unroll
  for (int i = 0; i < 2; ++i)
#pragma unroll
    for (int j = 0; j < 4; ++j) {
      const int col = n0 + 16 * j + l15;
      const float b = bdt[col];
#pragma unroll
      for (int r = 0; r < 8; ++r) {
        const float v  = acc[i][j][r] + b;
        const float sp = (v > 20.0f) ? v : log1pf(__expf(v));
        delta[(size_t)(m0 + 16 * i + roff + r) * D_INNER + col] = sp;
      }
    }
}

// Selective scan over time, one thread per channel; fused gate epilogue -> y bf16.
// B/C rows (uniform over channels) are staged into LDS with CDNA5 async
// global->LDS copies (ASYNCcnt), double-buffered across 8-step chunks.
__global__ void scan_kernel(const float* __restrict__ delta,
                            const __bf16* __restrict__ u2,
                            const float* __restrict__ xdbl,
                            const float* __restrict__ Aneg,
                            const float* __restrict__ Dvec,
                            const float* __restrict__ xr,   // res = xr[:, D_INNER:]
                            __bf16* __restrict__ ybf) {
  const int d = blockIdx.x * blockDim.x + threadIdx.x;   // 0..D_INNER-1
  // [buf][step][0..15]=B, [16..31]=C
  __shared__ float BC[2][8][32];
  const int NC = DT_RANK + 2 * D_STATE; // 96
  float An[D_STATE];
#pragma unroll
  for (int n = 0; n < D_STATE; ++n) An[n] = Aneg[(size_t)d * D_STATE + n];
  const float Dv = Dvec[d];
  float h[D_STATE];
#pragma unroll
  for (int n = 0; n < D_STATE; ++n) h[n] = 0.0f;

  const int tid  = threadIdx.x;
  const int lrow = tid >> 5;     // 0..7
  const int lcol = tid & 31;     // 0..31  (16 B + 16 C)

  // per-thread LDS byte offsets of the two staging buffers (LDS aperture:
  // low 32 bits of the flat shared-pointer are the LDS byte address)
  const unsigned lds0 = (unsigned)(size_t)(&BC[0][lrow][lcol]);
  const unsigned lds1 = (unsigned)(size_t)(&BC[1][lrow][lcol]);

  // prologue: async-fill buffer 0 with chunk t0=0
  async_load_f32_to_lds(lds0, xdbl + (size_t)lrow * NC + DT_RANK + lcol);
  wait_asynccnt0();
  __syncthreads();

  for (int t0 = 0; t0 < SEQ_L; t0 += 8) {
    const int buf = (t0 >> 3) & 1;
    // prefetch next chunk into the other buffer while computing this one
    if (t0 + 8 < SEQ_L) {
      const unsigned ldsn = buf ? lds0 : lds1;
      async_load_f32_to_lds(ldsn,
                            xdbl + (size_t)(t0 + 8 + lrow) * NC + DT_RANK + lcol);
    }
#pragma unroll
    for (int i = 0; i < 8; ++i) {
      const int t = t0 + i;
      const float dlt = delta[(size_t)t * D_INNER + d];
      const float ub  = (float)u2[(size_t)t * D_INNER + d];
      const float du  = dlt * ub;
      float acc = 0.0f;
#pragma unroll
      for (int n = 0; n < D_STATE; ++n) {
        const float dA = __expf(dlt * An[n]);
        h[n] = fmaf(dA, h[n], du * BC[buf][i][n]);
        acc  = fmaf(h[n], BC[buf][i][16 + n], acc);
      }
      const float yv = acc + ub * Dv;
      const float r  = xr[(size_t)t * (2 * D_INNER) + D_INNER + d];
      const float sr = r / (1.0f + __expf(-r));
      ybf[(size_t)t * D_INNER + d] = (__bf16)(yv * sr);
    }
    wait_asynccnt0();
    __syncthreads();
  }
}

// GEMM4: y_bf (L x D_INNER) @ W_out + x residual -> out fp32 (L x D_MODEL)
__global__ void gemm_out_kernel(const __bf16* __restrict__ A,
                                const __bf16* __restrict__ Bt,
                                const float* __restrict__ xin,
                                float* __restrict__ out) {
  int m0, n0;
  if (!block_coords_2x4(SEQ_L, D_MODEL, m0, n0)) return;
  v8f acc[2][4];
  wmma_block_2x4(A, Bt, D_INNER, m0, n0, acc);
  const int lane = threadIdx.x & 31;
  const int roff = (lane >> 4) * 8;
  const int l15  = lane & 15;
#pragma unroll
  for (int i = 0; i < 2; ++i)
#pragma unroll
    for (int j = 0; j < 4; ++j) {
      const int col = n0 + 16 * j + l15;
#pragma unroll
      for (int r = 0; r < 8; ++r) {
        const size_t idx = (size_t)(m0 + 16 * i + roff + r) * D_MODEL + col;
        out[idx] = acc[i][j][r] + xin[idx];
      }
    }
}

// ---------------- host launcher ----------------
static inline size_t align256(size_t x) { return (x + 255) & ~(size_t)255; }

extern "C" void kernel_launch(void* const* d_in, const int* in_sizes, int n_in,
                              void* d_out, int out_size, void* d_ws, size_t ws_size,
                              hipStream_t stream) {
  (void)in_sizes; (void)n_in; (void)out_size; (void)ws_size;
  const float* x      = (const float*)d_in[0];
  const float* W_in   = (const float*)d_in[1];
  const float* W_conv = (const float*)d_in[2];
  const float* b_conv = (const float*)d_in[3];
  const float* W_x    = (const float*)d_in[4];
  const float* W_dt   = (const float*)d_in[5];
  const float* b_dt   = (const float*)d_in[6];
  const float* A_log  = (const float*)d_in[7];
  const float* Dvec   = (const float*)d_in[8];
  const float* W_out  = (const float*)d_in[9];
  const float* ln_s   = (const float*)d_in[10];
  const float* ln_b   = (const float*)d_in[11];
  float* out = (float*)d_out;

  // workspace carve-out (all 256B aligned)
  char* w = (char*)d_ws;
  size_t off = 0;
  auto carve = [&](size_t bytes) -> void* {
    void* p = w + off; off = align256(off + bytes); return p;
  };
  __bf16* xn_bf  = (__bf16*)carve((size_t)SEQ_L * D_MODEL * 2);
  __bf16* WinT   = (__bf16*)carve((size_t)(2 * D_INNER) * D_MODEL * 2);
  __bf16* WxT    = (__bf16*)carve((size_t)(DT_RANK + 2 * D_STATE) * D_INNER * 2);
  __bf16* WdtT   = (__bf16*)carve((size_t)D_INNER * DT_RANK * 2);
  __bf16* WoutT  = (__bf16*)carve((size_t)D_MODEL * D_INNER * 2);
  float*  Aneg   = (float*) carve((size_t)D_INNER * D_STATE * 4);
  float*  xr     = (float*) carve((size_t)SEQ_L * (2 * D_INNER) * 4);
  __bf16* u2bf   = (__bf16*)carve((size_t)SEQ_L * D_INNER * 2);
  float*  xdbl   = (float*) carve((size_t)SEQ_L * (DT_RANK + 2 * D_STATE) * 4);
  __bf16* dtbf   = (__bf16*)carve((size_t)SEQ_L * DT_RANK * 2);
  float*  delta  = (float*) carve((size_t)SEQ_L * D_INNER * 4);
  __bf16* ybf    = (__bf16*)carve((size_t)SEQ_L * D_INNER * 2);

  // 1. LayerNorm -> bf16
  ln_kernel<<<SEQ_L, 256, 0, stream>>>(x, ln_s, ln_b, xn_bf);

  // 2. weight convert + transpose (every call; deterministic)
  {
    int n = D_MODEL * 2 * D_INNER;
    transpose_bf16_kernel<<<(n + 255) / 256, 256, 0, stream>>>(W_in, WinT, D_MODEL, 2 * D_INNER);
    n = D_INNER * (DT_RANK + 2 * D_STATE);
    transpose_bf16_kernel<<<(n + 255) / 256, 256, 0, stream>>>(W_x, WxT, D_INNER, DT_RANK + 2 * D_STATE);
    n = DT_RANK * D_INNER;
    transpose_bf16_kernel<<<(n + 255) / 256, 256, 0, stream>>>(W_dt, WdtT, DT_RANK, D_INNER);
    n = D_INNER * D_MODEL;
    transpose_bf16_kernel<<<(n + 255) / 256, 256, 0, stream>>>(W_out, WoutT, D_INNER, D_MODEL);
    n = D_INNER * D_STATE;
    aneg_kernel<<<(n + 255) / 256, 256, 0, stream>>>(A_log, Aneg, n);
  }

  // 3. GEMM1: xr = xn @ W_in   (32x64 per wave: 64*64=4096 waves -> 512 blocks)
  gemm_in_kernel<<<(SEQ_L / 32) * (2 * D_INNER / 64) / 8, 256, 0, stream>>>(xn_bf, WinT, xr);

  // 4. conv + SiLU -> u2 (bf16)
  conv_silu_kernel<<<(SEQ_L * D_INNER) / 256, 256, 0, stream>>>(xr, W_conv, b_conv, u2bf);

  // 5. GEMM2: x_dbl = u2 @ W_x  (+ dt in bf16) -- narrow N=96, 1x1 tiles
  gemm_xdbl_kernel<<<(SEQ_L / 16) * ((DT_RANK + 2 * D_STATE) / 16) / 8, 256, 0, stream>>>(
      u2bf, WxT, xdbl, dtbf);

  // 6. GEMM3: delta = softplus(dt @ W_dt + b_dt)   (2048 waves -> 256 blocks)
  gemm_delta_kernel<<<(SEQ_L / 32) * (D_INNER / 64) / 8, 256, 0, stream>>>(
      dtbf, WdtT, b_dt, delta);

  // 7. selective scan + gating -> y (bf16), async double-buffered B/C staging
  scan_kernel<<<D_INNER / 256, 256, 0, stream>>>(delta, u2bf, xdbl, Aneg, Dvec, xr, ybf);

  // 8. GEMM4: out = y @ W_out + x   (1024 waves -> 128 blocks)
  gemm_out_kernel<<<(SEQ_L / 32) * (D_MODEL / 64) / 8, 256, 0, stream>>>(ybf, WoutT, x, out);
}
